// GGNN_54631984005708
// MI455X (gfx1250) — matrix operand order repaired and checked by reading, not compiled, not run on hardware
//
#include <hip/hip_runtime.h>

#define N_NODES 100000
#define E_EDGES 3200000
#define H 64
#define DIN 15
#define DOUT 12
#define NT 5
#define G_GRAPHS 256
#define STEPS 6
#define NTILES (N_NODES / 16)   // 6250, exact

typedef __attribute__((ext_vector_type(16))) _Float16 v16h;
typedef __attribute__((ext_vector_type(8)))  _Float16 v8h;
typedef __attribute__((ext_vector_type(8)))  float    v8f;

__device__ __forceinline__ int lane_id() { return (int)(threadIdx.x & 31); }

__device__ __forceinline__ float sigmoidf_(float x) { return 1.0f / (1.0f + __expf(-x)); }

// ---- WMMA fragment helpers (ISA 7.12.2 layouts, wave32) -------------------

// A: 16x32 f16, row-major source, ld in elements. Per lane: row = lane&15,
// K-halves: lanes 0-15 hold K = {0..7,16..23}, lanes 16-31 hold K = {8..15,24..31}.
__device__ __forceinline__ v16h load_A(const _Float16* __restrict__ X, int ld,
                                       int row0, int kbase) {
  int l = lane_id();
  const _Float16* p = X + (size_t)(row0 + (l & 15)) * ld + kbase + ((l >> 4) << 3);
  v8h lo = *(const v8h*)p;          // K = kbase+kg .. +7   (16B aligned)
  v8h hi = *(const v8h*)(p + 16);   // K = kbase+16+kg .. +7
  v16h a;
#pragma unroll
  for (int i = 0; i < 8; ++i) { a[i] = lo[i]; a[i + 8] = hi[i]; }
  return a;
}

// B fragment from pre-packed fragment-major buffer: one contiguous 32B load/lane.
// Packed layout: P[((tile*32 + lane)*16 + i)] = W[kt*32 + (lane>>4)*8 + (i&7) + (i>>3)*16][ct*16 + (lane&15)]
__device__ __forceinline__ v16h load_Bp(const _Float16* __restrict__ P, int tile) {
  return *(const v16h*)(P + (((size_t)tile * 32 + lane_id()) << 4));
}

__device__ __forceinline__ v8f wmma_f16(v16h a, v16h b, v8f c) {
  return __builtin_amdgcn_wmma_f32_16x16x32_f16(false, a, false, b, (short)0, c,
                                                false, false);
}

// C/D 16x16 f32: VGPR v, lane l -> M = v + 8*(l>=16), N = l&15
__device__ __forceinline__ void store_C(float* __restrict__ C, int ldc, int row0,
                                        int col0, v8f acc) {
  int l = lane_id();
  float* p = C + (size_t)(row0 + ((l >> 4) << 3)) * ldc + col0 + (l & 15);
#pragma unroll
  for (int v = 0; v < 8; ++v) p[(size_t)v * ldc] = acc[v];
}

__device__ __forceinline__ void store_Ch(_Float16* __restrict__ C, int ldc, int row0,
                                         int col0, v8f acc) {
  int l = lane_id();
  _Float16* p = C + (size_t)(row0 + ((l >> 4) << 3)) * ldc + col0 + (l & 15);
#pragma unroll
  for (int v = 0; v < 8; ++v) p[(size_t)v * ldc] = (_Float16)acc[v];
}

__device__ __forceinline__ v8f load_C(const float* __restrict__ C, int ldc, int row0,
                                      int col0) {
  int l = lane_id();
  const float* p = C + (size_t)(row0 + ((l >> 4) << 3)) * ldc + col0 + (l & 15);
  v8f c;
#pragma unroll
  for (int v = 0; v < 8; ++v) c[v] = p[(size_t)v * ldc];
  return c;
}

// ---- weight packing: f32 [K, Ncols] -> f16 fragment-major ------------------
// tile = kt * (Ncols/16) + ct ; per tile, lane l holds 16 contiguous halves:
// W[kt*32 + (l>>4)*8 + (i&7) + (i>>3)*16][ct*16 + (l&15)], i = 0..15.
__global__ void k_pack_b(const float* __restrict__ W, _Float16* __restrict__ P,
                         int Ncols, int total) {
  int tid = blockIdx.x * blockDim.x + threadIdx.x;
  if (tid >= total) return;
  int i    = tid & 15;
  int lane = (tid >> 4) & 31;
  int tile = tid >> 9;
  int nct  = Ncols >> 4;
  int kt   = tile / nct;
  int ct   = tile - kt * nct;
  int k = kt * 32 + ((lane >> 4) << 3) + (i & 7) + ((i >> 3) << 4);
  int n = ct * 16 + (lane & 15);
  P[tid] = (_Float16)W[k * Ncols + n];
}

// ---- small utility kernels ------------------------------------------------

__global__ void k_deg(const int* __restrict__ dstI, float* __restrict__ deg) {
  int e = blockIdx.x * blockDim.x + threadIdx.x;
  if (e < E_EDGES) atomicAdd(&deg[dstI[e]], 1.0f);
}

__global__ void k_invdeg(float* __restrict__ deg) {
  int n = blockIdx.x * blockDim.x + threadIdx.x;
  if (n < N_NODES) deg[n] = 1.0f / fmaxf(deg[n], 1.0f);
}

// out0 = relu(x @ lin0_w + b); h = out0; also f16 shadows
__global__ void k_lin0(const float* __restrict__ x, const float* __restrict__ w,
                       const float* __restrict__ b, float* __restrict__ out0,
                       _Float16* __restrict__ o0h, float* __restrict__ h,
                       _Float16* __restrict__ hh) {
  int tid = blockIdx.x * blockDim.x + threadIdx.x;
  if (tid >= N_NODES * H) return;
  int n = tid >> 6, c = tid & 63;
  float acc = b[c];
#pragma unroll
  for (int k = 0; k < DIN; ++k) acc = fmaf(x[n * DIN + k], w[k * H + c], acc);
  acc = fmaxf(acc, 0.0f);
  out0[tid] = acc;
  o0h[tid] = (_Float16)acc;
  h[tid] = acc;
  hh[tid] = (_Float16)acc;
}

// ---- per-step kernels -----------------------------------------------------

// T[n, t*64+c] = (h @ W_t)[n, c], all NT types. One wave per 16-row tile.
// Pemb: NT packed 64x64 matrices (8 tiles each: tile = kc*4 + ct).
__global__ __launch_bounds__(256) void k_transform(const _Float16* __restrict__ hh,
                                                   const _Float16* __restrict__ Pemb,
                                                   float* __restrict__ T) {
  int wave = (int)((blockIdx.x * blockDim.x + threadIdx.x) >> 5);
  if (wave >= NTILES) return;
  int row0 = wave * 16;
  v16h a0 = load_A(hh, H, row0, 0);
  v16h a1 = load_A(hh, H, row0, 32);
#pragma unroll
  for (int t = 0; t < NT; ++t) {
    const _Float16* Pt = Pemb + t * H * H;
#pragma unroll
    for (int ct = 0; ct < 4; ++ct) {
      v8f acc = {};
      acc = wmma_f16(a0, load_Bp(Pt, 0 * 4 + ct), acc);
      acc = wmma_f16(a1, load_Bp(Pt, 1 * 4 + ct), acc);
      store_C(T, NT * H, row0, t * H + ct * 16, acc);
    }
  }
}

// Gather msg = T[src, attr] and scatter-add into agg[dst]. 16 lanes x float4 per edge.
// T (128 MB) and agg (25.6 MB) are L2-resident on MI455X (192 MB L2).
__global__ void k_edge(const int* __restrict__ srcI, const int* __restrict__ dstI,
                       const int* __restrict__ attr, const float* __restrict__ T,
                       float* __restrict__ agg) {
  unsigned gid = blockIdx.x * blockDim.x + threadIdx.x;
  if (gid >= (unsigned)E_EDGES * 16u) return;
  int e = (int)(gid >> 4);
  int c4 = ((int)gid & 15) * 4;
  int s = srcI[e], d = dstI[e], a = attr[e];
  const float4 v = *(const float4*)(T + (size_t)s * (NT * H) + a * H + c4);
  float* p = agg + (size_t)d * H + c4;
  atomicAdd(p + 0, v.x);
  atomicAdd(p + 1, v.y);
  atomicAdd(p + 2, v.z);
  atomicAdd(p + 3, v.w);
}

// m = relu(agg * invdeg + conv_bias), write f16 for WMMA A-operand
__global__ void k_mprep(const float* __restrict__ agg, const float* __restrict__ invdeg,
                        const float* __restrict__ cbias, _Float16* __restrict__ mh) {
  int tid = blockIdx.x * blockDim.x + threadIdx.x;
  if (tid >= N_NODES * H) return;
  int n = tid >> 6, c = tid & 63;
  float v = fmaxf(fmaf(agg[tid], invdeg[n], cbias[c]), 0.0f);
  mh[tid] = (_Float16)v;
}

// Fused GRU: gx/gh accumulated in WMMA fragments, gate math elementwise on
// fragments (r,z,n columns share the C/D lane->(M,N) map), h updated in place.
// Pih/Phh: packed 64x192 (nct=12, tile = kc*12 + col/16).
__global__ __launch_bounds__(256) void k_gru(const _Float16* __restrict__ mh,
                                             _Float16* __restrict__ hh,
                                             const _Float16* __restrict__ Pih,
                                             const _Float16* __restrict__ Phh,
                                             const float* __restrict__ bih,
                                             const float* __restrict__ bhh,
                                             float* __restrict__ h) {
  int wave = (int)((blockIdx.x * blockDim.x + threadIdx.x) >> 5);
  if (wave >= NTILES) return;
  int row0 = wave * 16;
  v16h am0 = load_A(mh, H, row0, 0);
  v16h am1 = load_A(mh, H, row0, 32);
  v16h ah0 = load_A(hh, H, row0, 0);   // read before in-place f16 update
  v16h ah1 = load_A(hh, H, row0, 32);
  int l15 = lane_id() & 15;
#pragma unroll
  for (int jt = 0; jt < 4; ++jt) {
    int cr = jt * 16, cz = 64 + jt * 16, cn = 128 + jt * 16;
    int tr = jt, tz = 4 + jt, tn = 8 + jt;      // column-tile indices (nct = 12)
    v8f xr = {}, xz = {}, xn = {}, hr = {}, hz = {}, hn = {};
    xr = wmma_f16(am0, load_Bp(Pih, 0 * 12 + tr), xr);
    xr = wmma_f16(am1, load_Bp(Pih, 1 * 12 + tr), xr);
    xz = wmma_f16(am0, load_Bp(Pih, 0 * 12 + tz), xz);
    xz = wmma_f16(am1, load_Bp(Pih, 1 * 12 + tz), xz);
    xn = wmma_f16(am0, load_Bp(Pih, 0 * 12 + tn), xn);
    xn = wmma_f16(am1, load_Bp(Pih, 1 * 12 + tn), xn);
    hr = wmma_f16(ah0, load_Bp(Phh, 0 * 12 + tr), hr);
    hr = wmma_f16(ah1, load_Bp(Phh, 1 * 12 + tr), hr);
    hz = wmma_f16(ah0, load_Bp(Phh, 0 * 12 + tz), hz);
    hz = wmma_f16(ah1, load_Bp(Phh, 1 * 12 + tz), hz);
    hn = wmma_f16(ah0, load_Bp(Phh, 0 * 12 + tn), hn);
    hn = wmma_f16(ah1, load_Bp(Phh, 1 * 12 + tn), hn);
    float bxr = bih[cr + l15], bxz = bih[cz + l15], bxn = bih[cn + l15];
    float bhr = bhh[cr + l15], bhz = bhh[cz + l15], bhn = bhh[cn + l15];
    v8f hold = load_C(h, H, row0, jt * 16);
    v8f hnew;
#pragma unroll
    for (int v = 0; v < 8; ++v) {
      float r  = sigmoidf_((xr[v] + bxr) + (hr[v] + bhr));
      float z  = sigmoidf_((xz[v] + bxz) + (hz[v] + bhz));
      float nn = tanhf((xn[v] + bxn) + r * (hn[v] + bhn));
      hnew[v] = (1.0f - z) * nn + z * hold[v];
    }
    store_C(h, H, row0, jt * 16, hnew);
    store_Ch(hh, H, row0, jt * 16, hnew);
  }
}

// Readout: WMMA stage-1 (K=128 concat-GEMM + K=64 j-GEMM) into LDS,
// VALU stage-2 (64 -> 12) + atomic segment-sum into [G, DOUT].
// PiW1: packed 128x64 (nct=4, tile = kt*4 + ct); PjW1: packed 64x64.
__global__ __launch_bounds__(32) void k_readout(
    const _Float16* __restrict__ hh, const _Float16* __restrict__ o0h,
    const _Float16* __restrict__ PiW1, const _Float16* __restrict__ PjW1,
    const float* __restrict__ i_b1, const float* __restrict__ i_w2,
    const float* __restrict__ i_b2, const float* __restrict__ j_b1,
    const float* __restrict__ j_w2, const float* __restrict__ j_b2,
    const int* __restrict__ batch, float* __restrict__ out) {
  __shared__ float s_i[16][H];
  __shared__ float s_j[16][H];
  int row0 = blockIdx.x * 16;
  v16h ah0 = load_A(hh, H, row0, 0);
  v16h ah1 = load_A(hh, H, row0, 32);
  v16h ao0 = load_A(o0h, H, row0, 0);
  v16h ao1 = load_A(o0h, H, row0, 32);
  int l = lane_id();
  int l15 = l & 15;
  int mb = (l >> 4) << 3;
#pragma unroll
  for (int ct = 0; ct < 4; ++ct) {
    int c0 = ct * 16;
    v8f ai = {};                                   // cat(h, out0) @ i_w1 (K=128)
    ai = wmma_f16(ah0, load_Bp(PiW1, 0 * 4 + ct), ai);
    ai = wmma_f16(ah1, load_Bp(PiW1, 1 * 4 + ct), ai);
    ai = wmma_f16(ao0, load_Bp(PiW1, 2 * 4 + ct), ai);
    ai = wmma_f16(ao1, load_Bp(PiW1, 3 * 4 + ct), ai);
    float bi = i_b1[c0 + l15];
#pragma unroll
    for (int v = 0; v < 8; ++v) s_i[mb + v][c0 + l15] = sigmoidf_(ai[v] + bi);
    v8f aj = {};                                   // h @ j_w1 (K=64)
    aj = wmma_f16(ah0, load_Bp(PjW1, 0 * 4 + ct), aj);
    aj = wmma_f16(ah1, load_Bp(PjW1, 1 * 4 + ct), aj);
    float bj = j_b1[c0 + l15];
#pragma unroll
    for (int v = 0; v < 8; ++v) s_j[mb + v][c0 + l15] = sigmoidf_(aj[v] + bj);
  }
  __syncthreads();
#pragma unroll
  for (int q = 0; q < 6; ++q) {              // 16 nodes x 12 outs = 192 tasks
    int tt = l + 32 * q;
    int node = tt / DOUT, d = tt % DOUT;
    float ai = i_b2[d], aj = j_b2[d];
    for (int k = 0; k < H; ++k) {
      ai = fmaf(s_i[node][k], i_w2[k * DOUT + d], ai);
      aj = fmaf(s_j[node][k], j_w2[k * DOUT + d], aj);
    }
    float gated = sigmoidf_(ai) * aj;
    atomicAdd(&out[batch[row0 + node] * DOUT + d], gated);
  }
}

// ---- host launch ----------------------------------------------------------

extern "C" void kernel_launch(void* const* d_in, const int* in_sizes, int n_in,
                              void* d_out, int out_size, void* d_ws, size_t ws_size,
                              hipStream_t stream) {
  const float* x        = (const float*)d_in[0];
  const int*   ei       = (const int*)d_in[1];
  const int*   eattr    = (const int*)d_in[2];
  const int*   batch    = (const int*)d_in[3];
  const float* lin0_w   = (const float*)d_in[4];
  const float* lin0_b   = (const float*)d_in[5];
  const float* eemb     = (const float*)d_in[6];
  const float* cbias    = (const float*)d_in[7];
  const float* w_ih     = (const float*)d_in[8];
  const float* w_hh     = (const float*)d_in[9];
  const float* b_ih     = (const float*)d_in[10];
  const float* b_hh     = (const float*)d_in[11];
  const float* i_w1     = (const float*)d_in[12];
  const float* i_b1     = (const float*)d_in[13];
  const float* i_w2     = (const float*)d_in[14];
  const float* i_b2     = (const float*)d_in[15];
  const float* j_w1     = (const float*)d_in[16];
  const float* j_b1     = (const float*)d_in[17];
  const float* j_w2     = (const float*)d_in[18];
  const float* j_b2     = (const float*)d_in[19];
  float* out = (float*)d_out;

  const int* srcI = ei;             // edge_index[0]
  const int* dstI = ei + E_EDGES;   // edge_index[1]

  char* base = (char*)d_ws;
  size_t off = 0;
  auto alloc = [&](size_t bytes) -> char* {
    char* p = base + off;
    off = (off + bytes + 255) & ~(size_t)255;
    return p;
  };
  _Float16* Pemb   = (_Float16*)alloc((size_t)NT * H * H * 2);
  _Float16* Pih    = (_Float16*)alloc((size_t)H * 3 * H * 2);
  _Float16* Phh    = (_Float16*)alloc((size_t)H * 3 * H * 2);
  _Float16* PiW1   = (_Float16*)alloc((size_t)2 * H * H * 2);
  _Float16* PjW1   = (_Float16*)alloc((size_t)H * H * 2);
  float*    invdeg = (float*)alloc((size_t)N_NODES * 4);
  float*    out0   = (float*)alloc((size_t)N_NODES * H * 4);
  _Float16* o0h    = (_Float16*)alloc((size_t)N_NODES * H * 2);
  float*    h      = (float*)alloc((size_t)N_NODES * H * 4);
  _Float16* hh     = (_Float16*)alloc((size_t)N_NODES * H * 2);
  _Float16* mh     = (_Float16*)alloc((size_t)N_NODES * H * 2);
  float*    agg    = (float*)alloc((size_t)N_NODES * H * 4);
  float*    T      = (float*)alloc((size_t)N_NODES * NT * H * 4);
  (void)ws_size; (void)n_in; (void)in_sizes; (void)out_size;

  // pack weights into WMMA fragment-major f16 (tiny)
  for (int t = 0; t < NT; ++t)
    k_pack_b<<<(H * H + 255) / 256, 256, 0, stream>>>(eemb + t * H * H,
                                                      Pemb + t * H * H, H, H * H);
  k_pack_b<<<(H * 3 * H + 255) / 256, 256, 0, stream>>>(w_ih, Pih, 3 * H, H * 3 * H);
  k_pack_b<<<(H * 3 * H + 255) / 256, 256, 0, stream>>>(w_hh, Phh, 3 * H, H * 3 * H);
  k_pack_b<<<(2 * H * H + 255) / 256, 256, 0, stream>>>(i_w1, PiW1, H, 2 * H * H);
  k_pack_b<<<(H * H + 255) / 256, 256, 0, stream>>>(j_w1, PjW1, H, H * H);

  // degree -> 1/max(deg,1)
  hipMemsetAsync(invdeg, 0, (size_t)N_NODES * 4, stream);
  k_deg<<<(E_EDGES + 255) / 256, 256, 0, stream>>>(dstI, invdeg);
  k_invdeg<<<(N_NODES + 255) / 256, 256, 0, stream>>>(invdeg);

  // lin0
  k_lin0<<<(N_NODES * H + 255) / 256, 256, 0, stream>>>(x, lin0_w, lin0_b, out0, o0h,
                                                        h, hh);

  const int waveBlocks = (NTILES * 32 + 255) / 256;  // 8 waves/block, 1 tile/wave
  for (int s = 0; s < STEPS; ++s) {
    k_transform<<<waveBlocks, 256, 0, stream>>>(hh, Pemb, T);
    hipMemsetAsync(agg, 0, (size_t)N_NODES * H * 4, stream);
    k_edge<<<(E_EDGES * 16 + 255) / 256, 256, 0, stream>>>(srcI, dstI, eattr, T, agg);
    k_mprep<<<(N_NODES * H + 255) / 256, 256, 0, stream>>>(agg, invdeg, cbias, mh);
    k_gru<<<waveBlocks, 256, 0, stream>>>(mh, hh, Pih, Phh, b_ih, b_hh, h);
  }

  hipMemsetAsync(out, 0, (size_t)G_GRAPHS * DOUT * 4, stream);
  k_readout<<<NTILES, 32, 0, stream>>>(hh, o0h, PiW1, PjW1, i_b1, i_w2, i_b2, j_b1,
                                       j_w2, j_b2, batch, out);
}
